// SwitchLinear_6854767805199
// MI455X (gfx1250) — matrix operand order repaired and checked by reading, not compiled, not run on hardware
//
#include <hip/hip_runtime.h>
#include <hip/hip_bf16.h>
#include <stdint.h>

// ---- problem constants (match reference) ----
#define TKN   2304      // B*N
#define MMDIM 1024
#define CDIM  2048
#define EXP   4
#define HDIM  8192
#define LN_EPS 1e-5f

typedef __bf16 bf16x16 __attribute__((ext_vector_type(16)));
typedef float  f32x8   __attribute__((ext_vector_type(8)));

union AFrag { uint32_t u[8]; bf16x16 v; };
union BFrag { uint32_t u[8]; bf16x16 v; };
union Pack2 { __bf16 h[2]; uint32_t u; };
union Pack4 { uint32_t pu[4]; uint4 q; };

__device__ __forceinline__ float gelu_exact(float x) {
    return 0.5f * x * (1.0f + erff(x * 0.70710678118654752440f));
}

// ---------------------------------------------------------------------------
// init: zero per-expert counters and the counts/route_prob_sum tail of d_out
// ---------------------------------------------------------------------------
__global__ void moe_init_kernel(int* devCnt, float* outTail) {
    int t = threadIdx.x;
    if (t < EXP) devCnt[t] = 0;
    if (t < 2 * EXP) outTail[t] = 0.0f;
}

// ---------------------------------------------------------------------------
// LayerNorm: one block per token, 1024 elems, write bf16 normalized activations
// ---------------------------------------------------------------------------
__global__ void moe_ln_kernel(const float* __restrict__ x,
                              const float* __restrict__ g,
                              const float* __restrict__ b,
                              __bf16* __restrict__ xn) {
    const int token = blockIdx.x;
    const int tid   = threadIdx.x;
    const float* xr = x + (size_t)token * MMDIM;

    float vals[4];
    float s = 0.f, s2 = 0.f;
#pragma unroll
    for (int i = 0; i < 4; ++i) {
        float v = xr[tid + i * 256];
        vals[i] = v; s += v; s2 += v * v;
    }
    __shared__ float rs[256], rs2[256];
    rs[tid] = s; rs2[tid] = s2;
    __syncthreads();
    for (int st = 128; st > 0; st >>= 1) {
        if (tid < st) { rs[tid] += rs[tid + st]; rs2[tid] += rs2[tid + st]; }
        __syncthreads();
    }
    const float mean = rs[0] * (1.0f / MMDIM);
    const float var  = rs2[0] * (1.0f / MMDIM) - mean * mean;
    const float rstd = rsqrtf(var + LN_EPS);
#pragma unroll
    for (int i = 0; i < 4; ++i) {
        int c = tid + i * 256;
        float v = (vals[i] - mean) * rstd * g[c] + b[c];
        xn[(size_t)token * MMDIM + c] = (__bf16)v;
    }
}

// ---------------------------------------------------------------------------
// Generic WMMA bf16 GEMM:  C[Meff,N] = act( gather(A)[Meff,K] * toBF16(B[K,N]) + bias )
//  - block = 256 threads = 8 waves; block tile 64(M) x 64(N); K step 32
//  - wave tile 16(M) x 32(N) -> 2x v_wmma_f32_16x16x32_bf16 per K step
//  - M tile on blockIdx.x (fast-varying) so concurrent blocks share the same
//    weight tiles through L2 (weights are the bandwidth-critical stream)
//  - A: bf16 row-major (lda), rows gathered through perm; per-thread row
//    pointer hoisted out of the K loop; b128 loads
//  - B: fp32 row-major weights (ldb), converted to bf16 while staging in LDS
// ---------------------------------------------------------------------------
__global__ void moe_gemm_wmma_kernel(const __bf16* __restrict__ A, int lda,
                                     const float*  __restrict__ B, long long strideB, int ldb,
                                     const float*  __restrict__ bias, int strideBias,
                                     float*  Cf, int ldcf,
                                     __bf16* Cb, int ldcb,
                                     const int* __restrict__ perm, int permCap,
                                     const int* __restrict__ cntPtr,
                                     int M, int N, int K, int applyGelu) {
    const int e = blockIdx.z;
    const float* Bp    = B    + (long long)e * strideB;
    const float* biasp = bias + (long long)e * strideBias;
    const int*   permp = perm ? (perm + (long long)e * permCap) : nullptr;
    const int Meff = cntPtr ? cntPtr[e] : M;

    const int mBase = blockIdx.x * 64;      // M fast-varying -> L2 weight sharing
    if (mBase >= Meff) return;              // block past this expert's token count
    const int nBase = blockIdx.y * 64;

    const int lane = threadIdx.x & 31;
    const int wid  = threadIdx.x >> 5;      // 8 waves (wave32)
    const int wm   = wid & 3;               // 0..3 -> M sub-tile (16 rows)
    const int wn   = wid >> 2;              // 0..1 -> N sub-tile (32 cols)

    __shared__ uint32_t sA[64][20];         // 64 rows x 32 bf16 (16 dwords) pad->20
    __shared__ uint32_t sB[32][36];         // 32 rows x 64 bf16 (32 dwords) pad->36

    // ---- per-thread A staging row: pointer computed once (loop-invariant) ----
    const int arow = threadIdx.x >> 2;      // 0..63
    const int aseg = threadIdx.x & 3;       // 4 dwords (8 bf16) each
    int rgA = -1;
    {
        int mrowA = mBase + arow;
        if (mrowA < Meff) rgA = permp ? permp[mrowA] : mrowA;
    }
    const __bf16* aBase = A + (size_t)(rgA < 0 ? 0 : rgA) * lda;

    // ---- B staging coords: row = K line, 8 consecutive floats per thread ----
    const int brow = threadIdx.x >> 3;      // 0..31
    const int bc0  = (threadIdx.x & 7) * 8; // 0..56

    f32x8 acc0 = {}, acc1 = {};

    for (int k0 = 0; k0 < K; k0 += 32) {
        // ---- stage A tile: one b128 per thread ----
        uint4 aw = make_uint4(0u, 0u, 0u, 0u);
        if (rgA >= 0)
            aw = reinterpret_cast<const uint4*>(aBase + k0)[aseg];
        *reinterpret_cast<uint4*>(&sA[arow][aseg * 4]) = aw;

        // ---- stage B tile: 8 fp32 -> 8 bf16, one b128 LDS store ----
        {
            const float* src = Bp + (size_t)(k0 + brow) * ldb + nBase + bc0;
            if (k0 + 32 < K) __builtin_prefetch(src + (size_t)32 * ldb, 0, 1);
            float4 f0 = *reinterpret_cast<const float4*>(src);
            float4 f1 = *reinterpret_cast<const float4*>(src + 4);
            float fv[8] = {f0.x, f0.y, f0.z, f0.w, f1.x, f1.y, f1.z, f1.w};
            Pack4 pk4;
#pragma unroll
            for (int j = 0; j < 4; ++j) {
                Pack2 p;
                p.h[0] = (__bf16)fv[2 * j];
                p.h[1] = (__bf16)fv[2 * j + 1];
                pk4.pu[j] = p.u;
            }
            *reinterpret_cast<uint4*>(&sB[brow][bc0 >> 1]) = pk4.q;
        }
        __syncthreads();

        // ---- fragments per ISA 7.12.2 layouts ----
        AFrag a; BFrag b0, b1;
        const int r = lane & 15, half = lane >> 4;
#pragma unroll
        for (int v = 0; v < 8; ++v) {
            // lanes 0-15: K pairs {0..3, 8..11}; lanes 16-31: {4..7, 12..15}
            int pair = (v < 4) ? (v + 4 * half) : (v + 4 + 4 * half);
            a.u[v] = sA[wm * 16 + r][pair];
        }
#pragma unroll
        for (int j = 0; j < 8; ++j) {        // lane = N column; K pairs in A's order
            int kk = (j < 4) ? (2 * j + 8 * half) : (2 * j + 8 + 8 * half);
            const __bf16* s0 = reinterpret_cast<const __bf16*>(&sB[kk][0]);
            const __bf16* s1 = reinterpret_cast<const __bf16*>(&sB[kk + 1][0]);
            Pack2 p0, p1;
            p0.h[0] = s0[wn * 32 + r];      p0.h[1] = s1[wn * 32 + r];
            p1.h[0] = s0[wn * 32 + 16 + r]; p1.h[1] = s1[wn * 32 + 16 + r];
            b0.u[j] = p0.u;
            b1.u[j] = p1.u;
        }
        acc0 = __builtin_amdgcn_wmma_f32_16x16x32_bf16(
                   false, a.v, false, b0.v, (short)0, acc0, false, false);
        acc1 = __builtin_amdgcn_wmma_f32_16x16x32_bf16(
                   false, a.v, false, b1.v, (short)0, acc1, false, false);
        __syncthreads();
    }

    // ---- epilogue: bias + optional exact GELU, scatter rows via perm ----
    const int half = lane >> 4;
    int  rowG[8];
    bool rowOk[8];
#pragma unroll
    for (int v = 0; v < 8; ++v) {
        int mrow = mBase + wm * 16 + v + 8 * half;  // C layout: VGPR v -> M = v + 8*half
        rowOk[v] = (mrow < Meff);
        rowG[v]  = rowOk[v] ? (permp ? permp[mrow] : mrow) : 0;
    }
#pragma unroll
    for (int s = 0; s < 2; ++s) {
        const int col = nBase + wn * 32 + s * 16 + (lane & 15);
        const float bv = biasp[col];
        const f32x8 accp = s ? acc1 : acc0;
#pragma unroll
        for (int v = 0; v < 8; ++v) {
            if (rowOk[v]) {
                float val = accp[v] + bv;
                if (applyGelu) val = gelu_exact(val);
                if (Cf) Cf[(size_t)rowG[v] * ldcf + col] = val;
                if (Cb) Cb[(size_t)rowG[v] * ldcb + col] = (__bf16)val;
            }
        }
    }
}

// ---------------------------------------------------------------------------
// Router: per-token softmax over E=4 logits, top-1 dispatch + stats
// ---------------------------------------------------------------------------
__global__ void moe_router_kernel(const float* __restrict__ t,
                                  const float* __restrict__ swW,   // [C,E] row-major
                                  const float* __restrict__ swb,
                                  int* __restrict__ devCnt,
                                  int* __restrict__ perm,          // [E, TKN]
                                  float* __restrict__ outTail) {   // counts[E], rps[E]
    const int token = blockIdx.x;
    const int tid   = threadIdx.x;
    const float* tr = t + (size_t)token * CDIM;

    float acc0 = 0.f, acc1 = 0.f, acc2 = 0.f, acc3 = 0.f;
    for (int c = tid; c < CDIM; c += 256) {
        float tv = tr[c];
        float4 w = *reinterpret_cast<const float4*>(swW + (size_t)c * EXP);
        acc0 += tv * w.x; acc1 += tv * w.y; acc2 += tv * w.z; acc3 += tv * w.w;
    }
    __shared__ float red[256][4];
    red[tid][0] = acc0; red[tid][1] = acc1; red[tid][2] = acc2; red[tid][3] = acc3;
    __syncthreads();
    for (int st = 128; st > 0; st >>= 1) {
        if (tid < st)
#pragma unroll
            for (int e = 0; e < EXP; ++e) red[tid][e] += red[tid + st][e];
        __syncthreads();
    }
    if (tid == 0) {
        float lg[EXP], mx = -3.4e38f;
#pragma unroll
        for (int e = 0; e < EXP; ++e) { lg[e] = red[0][e] + swb[e]; mx = fmaxf(mx, lg[e]); }
        float sum = 0.f;
#pragma unroll
        for (int e = 0; e < EXP; ++e) { lg[e] = __expf(lg[e] - mx); sum += lg[e]; }
        float inv = 1.0f / sum;
        int best = 0; float bestp = -1.f;
#pragma unroll
        for (int e = 0; e < EXP; ++e) {
            float p = lg[e] * inv;
            atomicAdd(&outTail[EXP + e], p);      // route_prob_sum
            if (p > bestp) { bestp = p; best = e; }
        }
        atomicAdd(&outTail[best], 1.0f);          // counts (as float)
        int pos = atomicAdd(&devCnt[best], 1);
        perm[best * TKN + pos] = token;
    }
}

// ---------------------------------------------------------------------------
extern "C" void kernel_launch(void* const* d_in, const int* in_sizes, int n_in,
                              void* d_out, int out_size, void* d_ws, size_t ws_size,
                              hipStream_t stream) {
    const float* x    = (const float*)d_in[0];
    const float* ln_g = (const float*)d_in[1];
    const float* ln_b = (const float*)d_in[2];
    const float* preW = (const float*)d_in[3];   // [MM, C]
    const float* preB = (const float*)d_in[4];   // [C]
    const float* swW  = (const float*)d_in[5];   // [C, E]
    const float* swB  = (const float*)d_in[6];   // [E]
    const float* W1   = (const float*)d_in[7];   // [E, C, H]
    const float* b1   = (const float*)d_in[8];   // [E, H]
    const float* W2   = (const float*)d_in[9];   // [E, H, C]
    const float* b2   = (const float*)d_in[10];  // [E, C]
    float* out = (float*)d_out;

    // workspace carve-up (256B aligned)
    char* ws = (char*)d_ws;
    size_t off = 0;
    auto carve = [&](size_t bytes) -> char* {
        char* p = ws + off;
        off = (off + bytes + 255) & ~(size_t)255;
        return p;
    };
    __bf16* xn   = (__bf16*)carve((size_t)TKN * MMDIM * 2);  // normalized, bf16
    float*  tF   = (float*) carve((size_t)TKN * CDIM  * 4);  // t fp32 (router)
    __bf16* tB   = (__bf16*)carve((size_t)TKN * CDIM  * 2);  // t bf16 (FFN A)
    __bf16* z    = (__bf16*)carve((size_t)TKN * HDIM  * 2);  // hidden, bf16 (by token row)
    int*    perm = (int*)   carve((size_t)EXP * TKN * 4);    // per-expert token lists
    int*    cnt  = (int*)   carve((size_t)EXP * 4);          // per-expert counters
    (void)ws_size; (void)in_sizes; (void)n_in; (void)out_size;

    float* outTail = out + (size_t)TKN * CDIM;   // counts[E] then route_prob_sum[E]

    moe_init_kernel<<<1, 64, 0, stream>>>(cnt, outTail);

    moe_ln_kernel<<<TKN, 256, 0, stream>>>(x, ln_g, ln_b, xn);

    // pre_linear + GELU: [TKN,1024] x [1024,2048] -> tF (fp32) and tB (bf16)
    moe_gemm_wmma_kernel<<<dim3(TKN / 64, CDIM / 64, 1), 256, 0, stream>>>(
        xn, MMDIM, preW, 0, CDIM, preB, 0,
        tF, CDIM, tB, CDIM,
        nullptr, 0, nullptr, TKN, CDIM, MMDIM, /*gelu=*/1);

    moe_router_kernel<<<TKN, 256, 0, stream>>>(tF, swW, swB, cnt, perm, outTail);

    // FFN layer 1 (gathered): z[token] = gelu( t[token] x W1[e] + b1[e] )
    moe_gemm_wmma_kernel<<<dim3(TKN / 64, HDIM / 64, EXP), 256, 0, stream>>>(
        tB, CDIM, W1, (long long)CDIM * HDIM, HDIM, b1, HDIM,
        nullptr, 0, z, HDIM,
        perm, TKN, cnt, TKN, HDIM, CDIM, /*gelu=*/1);

    // FFN layer 2 (gathered): out[token] = z[token] x W2[e] + b2[e]
    moe_gemm_wmma_kernel<<<dim3(TKN / 64, CDIM / 64, EXP), 256, 0, stream>>>(
        z, HDIM, W2, (long long)HDIM * CDIM, CDIM, b2, CDIM,
        out, CDIM, nullptr, 0,
        perm, TKN, cnt, TKN, CDIM, HDIM, /*gelu=*/0);
}
